// AttnBlock3D_70609262346811
// MI455X (gfx1250) — matrix operand order repaired and checked by reading, not compiled, not run on hardware
//
#include <hip/hip_runtime.h>
#include <hip/hip_bf16.h>

typedef __attribute__((ext_vector_type(16))) __bf16 v16bf;
typedef __attribute__((ext_vector_type(8)))  float  v8f;
typedef unsigned int u32x4 __attribute__((ext_vector_type(4)));
typedef int          i32x4 __attribute__((ext_vector_type(4)));
typedef int          i32x8 __attribute__((ext_vector_type(8)));

#define C512   512
#define PTOT   16384      // b*t*h*w tokens
#define NFRM   16
#define NTOK   1024       // tokens per frame (h*w)
#define NGRP   32
#define GSZ    (C512 / NGRP)   // 16 channels per group

#if defined(__has_builtin)
#if __has_builtin(__builtin_amdgcn_tensor_load_to_lds) && __has_builtin(__builtin_amdgcn_s_wait_tensorcnt)
#define USE_TDM 1
#endif
#endif

// ---------------------------------------------------------------------------
// GroupNorm stats: one block per group; group channels are contiguous in x.
// ---------------------------------------------------------------------------
__global__ __launch_bounds__(256)
void gn_stats_kernel(const float* __restrict__ x, float* __restrict__ meanrstd) {
    const int g = blockIdx.x;
    const float4* p4 = reinterpret_cast<const float4*>(x + (long)g * GSZ * PTOT);
    const int n4 = GSZ * PTOT / 4;
    float s = 0.f, ss = 0.f;
    for (int i = threadIdx.x; i < n4; i += 256) {
        float4 v = p4[i];
        s  += v.x + v.y + v.z + v.w;
        ss += v.x * v.x + v.y * v.y + v.z * v.z + v.w * v.w;
    }
    __shared__ float rs[256], rq[256];
    rs[threadIdx.x] = s; rq[threadIdx.x] = ss;
    __syncthreads();
    for (int off = 128; off > 0; off >>= 1) {
        if (threadIdx.x < off) {
            rs[threadIdx.x] += rs[threadIdx.x + off];
            rq[threadIdx.x] += rq[threadIdx.x + off];
        }
        __syncthreads();
    }
    if (threadIdx.x == 0) {
        const float inv_n = 1.0f / (float)(GSZ * PTOT);
        float mean = rs[0] * inv_n;
        float var  = rq[0] * inv_n - mean * mean;
        meanrstd[2 * g]     = mean;
        meanrstd[2 * g + 1] = rsqrtf(var + 1e-6f);
    }
}

// ---------------------------------------------------------------------------
// GN apply + transpose [c][p] -> token-major bf16 [p][c]; 32x32 LDS tiles.
// ---------------------------------------------------------------------------
__global__ __launch_bounds__(256)
void gn_apply_t_kernel(const float* __restrict__ x, const float* __restrict__ meanrstd,
                       const float* __restrict__ gamma, const float* __restrict__ beta,
                       __bf16* __restrict__ hbf) {
    __shared__ float tile[32][33];
    const int p0 = blockIdx.x * 32, c0 = blockIdx.y * 32;
    const int tx = threadIdx.x & 31, ty = threadIdx.x >> 5;   // ty 0..7
    #pragma unroll
    for (int i = 0; i < 4; ++i) {
        int cl = ty + i * 8;
        int c  = c0 + cl;
        int g  = c >> 4;
        float mean = meanrstd[2 * g], rstd = meanrstd[2 * g + 1];
        float v = x[(long)c * PTOT + p0 + tx];
        tile[cl][tx] = (v - mean) * rstd * gamma[c] + beta[c];
    }
    __syncthreads();
    #pragma unroll
    for (int i = 0; i < 4; ++i) {
        int pl = ty + i * 8;
        hbf[(long)(p0 + pl) * C512 + (c0 + tx)] = (__bf16)tile[tx][pl];
    }
}

// ---------------------------------------------------------------------------
// f32 -> bf16 weight pack
// ---------------------------------------------------------------------------
__global__ __launch_bounds__(256)
void pack_w_kernel(const float* __restrict__ w, __bf16* __restrict__ o, int n) {
    int i = blockIdx.x * 256 + threadIdx.x;
    if (i < n) o[i] = (__bf16)w[i];
}

// ---------------------------------------------------------------------------
// Row softmax over bf16 S, in place. One block per row of 1024.
// ---------------------------------------------------------------------------
__global__ __launch_bounds__(256)
void softmax_rows_kernel(__bf16* __restrict__ S) {
    __bf16* r = S + (long)blockIdx.x * NTOK;
    const int tid = threadIdx.x;
    float v[4];
    float m = -3.0e38f;
    #pragma unroll
    for (int i = 0; i < 4; ++i) { v[i] = (float)r[tid + i * 256]; m = fmaxf(m, v[i]); }
    __shared__ float red[256];
    red[tid] = m; __syncthreads();
    for (int off = 128; off > 0; off >>= 1) {
        if (tid < off) red[tid] = fmaxf(red[tid], red[tid + off]);
        __syncthreads();
    }
    m = red[0]; __syncthreads();
    float s = 0.f;
    #pragma unroll
    for (int i = 0; i < 4; ++i) { v[i] = __expf(v[i] - m); s += v[i]; }
    red[tid] = s; __syncthreads();
    for (int off = 128; off > 0; off >>= 1) {
        if (tid < off) red[tid] += red[tid + off];
        __syncthreads();
    }
    float inv = 1.0f / red[0];
    #pragma unroll
    for (int i = 0; i < 4; ++i) r[tid + i * 256] = (__bf16)(v[i] * inv);
}

#ifdef USE_TDM
// ---------------------------------------------------------------------------
// TDM: DMA one [128 rows x 32 cols] bf16 tile (row stride K elems) into LDS.
// LDS layout produced: row stride 40 bf16 (pad 4 DWORDs per 16 DWORDs),
// matching the padded LDS tiles the compute path reads.
// Descriptor per CDNA5 ISA (D# groups 0/1, groups 2/3 zero).
// ---------------------------------------------------------------------------
__device__ __forceinline__ void tdm_load_tile_128x32(unsigned int lds_off,
                                                     const void* gptr, int K) {
    unsigned long long ga = (unsigned long long)(uintptr_t)gptr;
    u32x4 g0 = {
        1u,                                    // count=1, user mode, no gather
        lds_off,                               // lds_addr [63:32]
        (unsigned int)ga,                      // global_addr [95:64]
        (unsigned int)((ga >> 32) & 0x01FFFFFFu) | 0x80000000u  // addr[56:32] | type=2
    };
    // group1: data_size=2B, pad_enable, pad_interval=16 DW, pad_amount=4 DW
    i32x8 g1 = {
        (int)0x06D10000u,                      // mask=0,ds=1,pad_en,int=3,amt=3
        (int)((K & 0xFFFF) << 16),             // tensor_dim0 lo16 (abar=0)
        (int)(((unsigned)K >> 16) | (128u << 16)), // tensor_dim0 hi / tensor_dim1 lo
        (int)(32 << 16),                       // tensor_dim1 hi=0, tile_dim0=32
        128,                                   // tile_dim1=128, tile_dim2=0
        K,                                     // tensor_dim0_stride lo32 (elems)
        0, 0                                   // strides hi / dim1_stride
    };
    i32x4 z4 = {0, 0, 0, 0};
#if defined(__clang_major__) && __clang_major__ >= 23
    i32x8 z8 = {0, 0, 0, 0, 0, 0, 0, 0};
    __builtin_amdgcn_tensor_load_to_lds(g0, g1, z4, z4, z8, 0);
#else
    __builtin_amdgcn_tensor_load_to_lds(g0, g1, z4, z4, 0);
#endif
}
#endif

// ---------------------------------------------------------------------------
// bf16 WMMA GEMM: C[M][N] = sum_k A[m][k] * Bt[n][k]   (A,Bt row-major bf16)
// Block tile 128x128, BK=32, 256 threads = 8 waves (4 along M x 2 along N),
// each wave computes 32x64 = 2x4 tiles of v_wmma_f32_16x16x32_bf16.
// Global->LDS via TDM double-buffer (wave 0 issues, TENSORcnt sync) when
// available; vector-load fallback otherwise.
// Epilogues:
//   0: bf16 out[m*N+n] + bias[n]                             (q,k)
//   1: bf16 transposed out[(m/FR*N+n)*FR+m%FR] + bias[n]     (v -> [f][c][j])
//   2: bf16 out[m*N+n] * scale                               (scores)
//   3: bf16 out[m*N+n]                                       (attn @ V)
//   4: f32  out[n*M+m] + bias[n] + resid[n*M+m]              (final, [c][p])
// ---------------------------------------------------------------------------
template <int EPI>
__global__ __launch_bounds__(256)
void gemm_bf16_wmma(const __bf16* __restrict__ A, const __bf16* __restrict__ Bt,
                    void* __restrict__ Out, const float* __restrict__ bias,
                    const float* __restrict__ resid,
                    int M, int N, int K, float scale, int FR,
                    long sA, long sB, long sO) {
    constexpr int BM = 128, BN = 128, BK = 32, BKP = 40;   // +8 pad = TDM pad cfg
    __shared__ __align__(16) __bf16 As[2][BM][BKP];
    __shared__ __align__(16) __bf16 Bs[2][BN][BKP];

    A  += (long)blockIdx.z * sA;
    Bt += (long)blockIdx.z * sB;

    const int tid  = threadIdx.x;
    const int wave = tid >> 5;
    const int lane = tid & 31;
    const int lh   = lane >> 4;       // lane half: 0 or 1
    const int l15  = lane & 15;
    const int mBase = (wave & 3) * 32;   // 4 waves along M
    const int nBase = (wave >> 2) * 64;  // 2 waves along N

    const __bf16* Ag = A  + (long)blockIdx.y * BM * K;
    const __bf16* Bg = Bt + (long)blockIdx.x * BN * K;

    v8f acc[2][4];
    #pragma unroll
    for (int mi = 0; mi < 2; ++mi)
        #pragma unroll
        for (int ni = 0; ni < 4; ++ni)
            acc[mi][ni] = v8f{0.f, 0.f, 0.f, 0.f, 0.f, 0.f, 0.f, 0.f};

    union Frag { uint4 q[2]; v16bf v; };
    const int ksteps = K / BK;

#ifdef USE_TDM
    const unsigned int ldsA0 = (unsigned int)(uintptr_t)&As[0][0][0];
    const unsigned int ldsA1 = (unsigned int)(uintptr_t)&As[1][0][0];
    const unsigned int ldsB0 = (unsigned int)(uintptr_t)&Bs[0][0][0];
    const unsigned int ldsB1 = (unsigned int)(uintptr_t)&Bs[1][0][0];
    if (wave == 0) {  // prologue: DMA tiles for kt=0 into buffer 0
        tdm_load_tile_128x32(ldsA0, Ag, K);
        tdm_load_tile_128x32(ldsB0, Bg, K);
    }
#endif

    for (int kt = 0; kt < ksteps; ++kt) {
        const int cur = kt & 1;
#ifdef USE_TDM
        if (wave == 0) {
            if (kt + 1 < ksteps) {   // DMA next tiles into the other buffer
                const int nxt = cur ^ 1;
                tdm_load_tile_128x32(nxt ? ldsA1 : ldsA0, Ag + (long)(kt + 1) * BK, K);
                tdm_load_tile_128x32(nxt ? ldsB1 : ldsB0, Bg + (long)(kt + 1) * BK, K);
                __builtin_amdgcn_s_wait_tensorcnt(2);  // tiles for kt landed
            } else {
                __builtin_amdgcn_s_wait_tensorcnt(0);
            }
        }
        __syncthreads();
#else
        // vector-load fallback: 128x32 bf16 tiles, 512 chunks of 8 bf16
        #pragma unroll
        for (int i = 0; i < 2; ++i) {
            int id  = tid + i * 256;
            int row = id >> 2;
            int cc  = (id & 3) * 8;
            *reinterpret_cast<uint4*>(&As[cur][row][cc]) =
                *reinterpret_cast<const uint4*>(Ag + (long)row * K + kt * BK + cc);
            *reinterpret_cast<uint4*>(&Bs[cur][row][cc]) =
                *reinterpret_cast<const uint4*>(Bg + (long)row * K + kt * BK + cc);
        }
        __syncthreads();
#endif
        // A frags: lane<16 -> M=l15, K{0..7,16..23}; lane>=16 -> K{8..15,24..31}
        Frag af[2];
        #pragma unroll
        for (int mi = 0; mi < 2; ++mi) {
            int r = mBase + mi * 16 + l15;
            int ko = lh * 8;
            af[mi].q[0] = *reinterpret_cast<const uint4*>(&As[cur][r][ko]);
            af[mi].q[1] = *reinterpret_cast<const uint4*>(&As[cur][r][ko + 16]);
        }
        // B frags: column N=l15, lane<16 -> K{0..7,16..23}; lane>=16 -> K{8..15,24..31}
        Frag bfm[4];
        #pragma unroll
        for (int ni = 0; ni < 4; ++ni) {
            int n = nBase + ni * 16 + l15;
            int ko = lh * 8;
            bfm[ni].q[0] = *reinterpret_cast<const uint4*>(&Bs[cur][n][ko]);
            bfm[ni].q[1] = *reinterpret_cast<const uint4*>(&Bs[cur][n][ko + 16]);
        }
        #pragma unroll
        for (int mi = 0; mi < 2; ++mi)
            #pragma unroll
            for (int ni = 0; ni < 4; ++ni)
                acc[mi][ni] = __builtin_amdgcn_wmma_f32_16x16x32_bf16(
                    false, af[mi].v, false, bfm[ni].v,
                    (short)0, acc[mi][ni], false, false);
        __syncthreads();
    }

    // Epilogue. C/D layout: VGPR v, lanes 0-15: M=m0+v,N=lane ; lanes16-31: M=m0+8+v
    #pragma unroll
    for (int mi = 0; mi < 2; ++mi) {
        #pragma unroll
        for (int ni = 0; ni < 4; ++ni) {
            #pragma unroll
            for (int v = 0; v < 8; ++v) {
                int m = blockIdx.y * BM + mBase + mi * 16 + lh * 8 + v;
                int n = blockIdx.x * BN + nBase + ni * 16 + l15;
                float val = acc[mi][ni][v];
                if constexpr (EPI == 0) {
                    __bf16* O = (__bf16*)Out + (long)blockIdx.z * sO;
                    O[(long)m * N + n] = (__bf16)(val + bias[n]);
                } else if constexpr (EPI == 1) {
                    __bf16* O = (__bf16*)Out;
                    int f = m / FR, rr = m % FR;
                    O[((long)f * N + n) * FR + rr] = (__bf16)(val + bias[n]);
                } else if constexpr (EPI == 2) {
                    __bf16* O = (__bf16*)Out + (long)blockIdx.z * sO;
                    O[(long)m * N + n] = (__bf16)(val * scale);
                } else if constexpr (EPI == 3) {
                    __bf16* O = (__bf16*)Out + (long)blockIdx.z * sO;
                    O[(long)m * N + n] = (__bf16)val;
                } else {  // EPI == 4: final, transposed f32 + bias + residual
                    float* O = (float*)Out;
                    long idx = (long)n * M + m;
                    O[idx] = val + bias[n] + resid[idx];
                }
            }
        }
    }
}

// ---------------------------------------------------------------------------
extern "C" void kernel_launch(void* const* d_in, const int* in_sizes, int n_in,
                              void* d_out, int out_size, void* d_ws, size_t ws_size,
                              hipStream_t stream) {
    const float* x     = (const float*)d_in[0];
    const float* gamma = (const float*)d_in[1];
    const float* beta  = (const float*)d_in[2];
    const float* wq    = (const float*)d_in[3];
    const float* bq    = (const float*)d_in[4];
    const float* wk    = (const float*)d_in[5];
    const float* bk    = (const float*)d_in[6];
    const float* wv    = (const float*)d_in[7];
    const float* bv    = (const float*)d_in[8];
    const float* wo    = (const float*)d_in[9];
    const float* bo    = (const float*)d_in[10];
    float* out = (float*)d_out;

    // workspace bump allocator (256B aligned)
    size_t off = 0;
    auto bump = [&](size_t bytes) -> void* {
        void* p = (char*)d_ws + off;
        off += (bytes + 255) & ~(size_t)255;
        return p;
    };
    float*  meanrstd = (float*)bump(NGRP * 2 * sizeof(float));
    __bf16* wqb  = (__bf16*)bump((size_t)C512 * C512 * 2);
    __bf16* wkb  = (__bf16*)bump((size_t)C512 * C512 * 2);
    __bf16* wvb  = (__bf16*)bump((size_t)C512 * C512 * 2);
    __bf16* wob  = (__bf16*)bump((size_t)C512 * C512 * 2);
    __bf16* hbf  = (__bf16*)bump((size_t)PTOT * C512 * 2);   // GN output, token-major
    __bf16* qbf  = (__bf16*)bump((size_t)PTOT * C512 * 2);   // [p][c]
    __bf16* kbf  = (__bf16*)bump((size_t)PTOT * C512 * 2);   // [p][c]
    __bf16* vtb  = (__bf16*)bump((size_t)PTOT * C512 * 2);   // [f][c][j]
    __bf16* Sbuf = (__bf16*)bump((size_t)NFRM * NTOK * NTOK * 2);  // scores
    __bf16* aout = (__bf16*)bump((size_t)PTOT * C512 * 2);   // attn output [p][c]
    (void)ws_size; (void)n_in; (void)in_sizes; (void)out_size;

    const float scale = 0.04419417382415922f;   // 512^-0.5

    // 1) GroupNorm stats
    gn_stats_kernel<<<NGRP, 256, 0, stream>>>(x, meanrstd);
    // 2) weight packs
    pack_w_kernel<<<(C512 * C512 + 255) / 256, 256, 0, stream>>>(wq, wqb, C512 * C512);
    pack_w_kernel<<<(C512 * C512 + 255) / 256, 256, 0, stream>>>(wk, wkb, C512 * C512);
    pack_w_kernel<<<(C512 * C512 + 255) / 256, 256, 0, stream>>>(wv, wvb, C512 * C512);
    pack_w_kernel<<<(C512 * C512 + 255) / 256, 256, 0, stream>>>(wo, wob, C512 * C512);
    // 3) GN apply + transpose to token-major bf16
    gn_apply_t_kernel<<<dim3(PTOT / 32, C512 / 32), 256, 0, stream>>>(x, meanrstd, gamma, beta, hbf);
    // 4) q,k projections  [16384,512] x [512,512]
    gemm_bf16_wmma<0><<<dim3(C512 / 128, PTOT / 128, 1), 256, 0, stream>>>(
        hbf, wqb, qbf, bq, nullptr, PTOT, C512, C512, 1.f, 0, 0, 0, 0);
    gemm_bf16_wmma<0><<<dim3(C512 / 128, PTOT / 128, 1), 256, 0, stream>>>(
        hbf, wkb, kbf, bk, nullptr, PTOT, C512, C512, 1.f, 0, 0, 0, 0);
    // 5) v projection stored per-frame transposed -> [f][c][j]
    gemm_bf16_wmma<1><<<dim3(C512 / 128, PTOT / 128, 1), 256, 0, stream>>>(
        hbf, wvb, vtb, bv, nullptr, PTOT, C512, C512, 1.f, NTOK, 0, 0, 0);
    // 6) S = scale * Q K^T per frame  [1024,512]x[512,1024], batch 16
    gemm_bf16_wmma<2><<<dim3(NTOK / 128, NTOK / 128, NFRM), 256, 0, stream>>>(
        qbf, kbf, Sbuf, nullptr, nullptr, NTOK, NTOK, C512, scale, 0,
        (long)NTOK * C512, (long)NTOK * C512, (long)NTOK * NTOK);
    // 7) softmax rows in place
    softmax_rows_kernel<<<NFRM * NTOK, 256, 0, stream>>>(Sbuf);
    // 8) out = P @ V per frame  [1024,1024]x[1024,512]
    gemm_bf16_wmma<3><<<dim3(C512 / 128, NTOK / 128, NFRM), 256, 0, stream>>>(
        Sbuf, vtb, aout, nullptr, nullptr, NTOK, C512, NTOK, 1.f, 0,
        (long)NTOK * NTOK, (long)C512 * NTOK, (long)NTOK * C512);
    // 9) final projection + bias + residual, f32 transposed store to [c][p]
    gemm_bf16_wmma<4><<<dim3(C512 / 128, PTOT / 128, 1), 256, 0, stream>>>(
        aout, wob, out, bo, x, PTOT, C512, C512, 1.f, PTOT, 0, 0, 0);
}
